// HeteroGATConv_58265526338119
// MI455X (gfx1250) — matrix operand order repaired and checked by reading, not compiled, not run on hardware
//
#include <hip/hip_runtime.h>
#include <hip/hip_bf16.h>

typedef __attribute__((ext_vector_type(16))) _Float16 v16h;
typedef __attribute__((ext_vector_type(8)))  _Float16 v8h;
typedef __attribute__((ext_vector_type(8)))  float    v8f;

#define HEADS 4
#define DIM   128
#define DHEAD 32

// ---------------- elementwise converts ----------------
__global__ void k_f32_to_f16(const float* __restrict__ in, _Float16* __restrict__ out, int n) {
  int i = blockIdx.x * blockDim.x + threadIdx.x;
  if (i < n) out[i] = (_Float16)in[i];
}

__global__ void k_relu_to_f16(const float* __restrict__ in, _Float16* __restrict__ out, int n) {
  int i = blockIdx.x * blockDim.x + threadIdx.x;
  if (i < n) { float v = in[i]; out[i] = (_Float16)(v > 0.f ? v : 0.f); }
}

// ---------------- W swizzle: [128k x 128n] f32 -> B fragments f16 ----------------
// Layout: [tileN(8)][kc(4)][lane(32)][j(16)] ; per lane: col n = tn*16+(lane&15),
// K = kc*32 + (lane>=16 ? 8 : 0) + (j&7) + (j>=8 ? 16 : 0)  (dense 32x16 f16 B layout, matches A table order)
__global__ void k_swizzle_w(const float* __restrict__ W, _Float16* __restrict__ Wsw) {
  int idx = blockIdx.x * blockDim.x + threadIdx.x;
  if (idx >= 8 * 4 * 32 * 16) return;
  int j    = idx & 15;
  int lane = (idx >> 4) & 31;
  int kc   = (idx >> 9) & 3;
  int tn   = idx >> 11;
  int n = tn * 16 + (lane & 15);
  int k = kc * 32 + ((lane >> 4) << 3) + (j & 7) + ((j >> 3) << 4);
  Wsw[idx] = (_Float16)W[k * DIM + n];
}

// ---------------- WMMA GEMM: Z[N,128] = Xh[N,128] @ W ----------------
// 256 threads = 8 waves; block owns a 16-row M tile; wave w owns N-tile w.
__global__ void k_gemm_wmma(const _Float16* __restrict__ X, const _Float16* __restrict__ Wsw,
                            float* __restrict__ Z, int nrows) {
  int wave = threadIdx.x >> 5;
  int lane = threadIdx.x & 31;
  int m0 = blockIdx.x << 4;
  int mrow = m0 + (lane & 15);
  if (mrow >= nrows) mrow = nrows - 1;          // duplicate-row clamp keeps EXEC all-ones
  int klo = (lane >> 4) << 3;                   // A layout: lanes>=16 hold K+8
  const _Float16* xrow = X + (size_t)mrow * DIM + klo;
  v8f acc = {};
#pragma unroll
  for (int kc = 0; kc < 4; ++kc) {
    union { v16h v; v8h h[2]; } A;
    A.h[0] = *(const v8h*)(xrow + kc * 32);        // K = kc*32 + klo + 0..7
    A.h[1] = *(const v8h*)(xrow + kc * 32 + 16);   // K = kc*32 + 16 + klo + 0..7
    v16h B = *(const v16h*)(Wsw + (((wave << 2) + kc) * 32 + lane) * 16);
    acc = __builtin_amdgcn_wmma_f32_16x16x32_f16(false, A.v, false, B,
                                                 (short)0, acc, false, false);
  }
  int col   = (wave << 4) + (lane & 15);
  int rbase = m0 + ((lane >> 4) << 3);          // lanes>=16 hold rows M=8..15
  if (m0 + 16 <= nrows) {                       // uniform (scalar) branch: full tile
    float* zp = Z + (size_t)rbase * DIM + col;
#pragma unroll
    for (int r = 0; r < 8; ++r) zp[(size_t)r * DIM] = acc[r];
  } else {                                      // tail tile (never taken when N%16==0)
#pragma unroll
    for (int r = 0; r < 8; ++r) {
      int row = rbase + r;
      if (row < nrows) Z[(size_t)row * DIM + col] = acc[r];
    }
  }
}

// ---------------- per-node attention projections ----------------
__global__ void k_eler(const float* __restrict__ Z, const float* __restrict__ al,
                       const float* __restrict__ ar, float* __restrict__ el,
                       float* __restrict__ er, int n) {  // n = N*HEADS
  int i = blockIdx.x * blockDim.x + threadIdx.x;
  if (i >= n) return;
  int h = i & (HEADS - 1);
  int node = i >> 2;
  const float* zp  = Z + (size_t)node * DIM + h * DHEAD;
  const float* alp = al + h * DHEAD;
  const float* arp = ar + h * DHEAD;
  float sl = 0.f, sr = 0.f;
#pragma unroll
  for (int d = 0; d < DHEAD; ++d) { float z = zp[d]; sl += z * alp[d]; sr += z * arp[d]; }
  el[i] = sl; er[i] = sr;
}

// ---------------- init: out = bias, m = 0(mapped -min), s = 0 ----------------
__global__ void k_init(float* __restrict__ out, const float* __restrict__ bias,
                       unsigned* __restrict__ m, float* __restrict__ s, int n) {  // n = N*128
  int i = blockIdx.x * blockDim.x + threadIdx.x;
  if (i >= n) return;
  int c = i & (DIM - 1);
  out[i] = bias[c];
  int node = i >> 7;
  if (c < HEADS)            m[node * HEADS + c] = 0u;
  else if (c < 2 * HEADS)   s[node * HEADS + (c - HEADS)] = 0.f;
}

// monotonic f32 <-> uint mapping (unsigned order == float order)
__device__ __forceinline__ unsigned fmap(float v) {
  unsigned u = __float_as_uint(v);
  return (u & 0x80000000u) ? ~u : (u | 0x80000000u);
}
__device__ __forceinline__ float funmap(unsigned u) {
  return __uint_as_float((u & 0x80000000u) ? (u & 0x7FFFFFFFu) : ~u);
}

// ---------------- edge pass 1: logits + segment max ----------------
__global__ void k_edge_max(const int* __restrict__ src, const int* __restrict__ dst,
                           const float* __restrict__ el, const float* __restrict__ er,
                           float* __restrict__ ebuf, unsigned* __restrict__ m, int n) { // n=E*H
  int i = blockIdx.x * blockDim.x + threadIdx.x;
  if (i >= n) return;
  int h = i & (HEADS - 1);
  int e = i >> 2;
  int sN = src[e], dN = dst[e];
  float v = el[sN * HEADS + h] + er[dN * HEADS + h];
  v = v > 0.f ? v : 0.2f * v;                  // leaky_relu(0.2)
  ebuf[i] = v;
  atomicMax(&m[dN * HEADS + h], fmap(v));
}

// ---------------- edge pass 2: exp + segment sum ----------------
__global__ void k_edge_expsum(const int* __restrict__ dst, float* __restrict__ ebuf,
                              const unsigned* __restrict__ m, float* __restrict__ s, int n) {
  int i = blockIdx.x * blockDim.x + threadIdx.x;
  if (i >= n) return;
  int h = i & (HEADS - 1);
  int e = i >> 2;
  int dN = dst[e];
  float ex = __expf(ebuf[i] - funmap(m[dN * HEADS + h]));
  ebuf[i] = ex;
  atomicAdd(&s[dN * HEADS + h], ex);
}

// ---------------- edge pass 3: normalized scatter-aggregate ----------------
// One thread per (edge, 4-channel group): 1 x b128 load of z, 4 atomic adds.
__global__ void k_edge_agg(const int* __restrict__ src, const int* __restrict__ dst,
                           const float* __restrict__ ebuf, const float* __restrict__ s,
                           const float* __restrict__ Z, float* __restrict__ out, long long n) {
  long long i = (long long)blockIdx.x * blockDim.x + threadIdx.x;
  if (i >= n) return;                           // n = E * 32
  int q = (int)(i & 31);                        // which float4 within the 128 channels
  int e = (int)(i >> 5);
  int c = q << 2;                               // channel base (16B aligned)
  int h = q >> 3;                               // 8 groups per 32-wide head
  int sN = src[e], dN = dst[e];
  float alpha = ebuf[e * HEADS + h] / s[dN * HEADS + h];
  const float4 zv = *(const float4*)(Z + (size_t)sN * DIM + c);
  float* op = out + (size_t)dN * DIM + c;
  atomicAdd(op + 0, zv.x * alpha);
  atomicAdd(op + 1, zv.y * alpha);
  atomicAdd(op + 2, zv.z * alpha);
  atomicAdd(op + 3, zv.w * alpha);
}

extern "C" void kernel_launch(void* const* d_in, const int* in_sizes, int n_in,
                              void* d_out, int out_size, void* d_ws, size_t ws_size,
                              hipStream_t stream) {
  const float* x   = (const float*)d_in[0];
  const float* W1  = (const float*)d_in[1];
  const float* al1 = (const float*)d_in[2];
  const float* ar1 = (const float*)d_in[3];
  const float* b1  = (const float*)d_in[4];
  const float* W2  = (const float*)d_in[5];
  const float* al2 = (const float*)d_in[6];
  const float* ar2 = (const float*)d_in[7];
  const float* b2  = (const float*)d_in[8];
  const int*   src = (const int*)d_in[9];
  const int*   dst = (const int*)d_in[10];
  const int N = in_sizes[0] / DIM;
  const int E = in_sizes[9];

  char* ws = (char*)d_ws;
  size_t off = 0;
  auto carve = [&](size_t bytes) -> char* {
    char* p = ws + off;
    off = (off + bytes + 255) & ~(size_t)255;
    return p;
  };
  _Float16* xh  = (_Float16*)carve((size_t)N * DIM * 2);
  _Float16* h16 = (_Float16*)carve((size_t)N * DIM * 2);
  _Float16* W1s = (_Float16*)carve((size_t)DIM * DIM * 2);
  _Float16* W2s = (_Float16*)carve((size_t)DIM * DIM * 2);
  float*    Z   = (float*)carve((size_t)N * DIM * 4);
  float*    H32 = (float*)carve((size_t)N * DIM * 4);
  float*    el  = (float*)carve((size_t)N * HEADS * 4);
  float*    er  = (float*)carve((size_t)N * HEADS * 4);
  unsigned* m   = (unsigned*)carve((size_t)N * HEADS * 4);
  float*    s   = (float*)carve((size_t)N * HEADS * 4);
  float*    eb  = (float*)carve((size_t)E * HEADS * 4);

  const int T = 256;
  const int nND = N * DIM, nNH = N * HEADS, nEH = E * HEADS;
  const long long nEG = (long long)E * 32;      // edge-aggregate work items (float4 groups)

  // prep
  k_f32_to_f16<<<(nND + T - 1) / T, T, 0, stream>>>(x, xh, nND);
  k_swizzle_w<<<(DIM * DIM + T - 1) / T, T, 0, stream>>>(W1, W1s);
  k_swizzle_w<<<(DIM * DIM + T - 1) / T, T, 0, stream>>>(W2, W2s);

  // ----- layer 1 -----
  k_gemm_wmma<<<(N + 15) / 16, 256, 0, stream>>>(xh, W1s, Z, N);
  k_eler<<<(nNH + T - 1) / T, T, 0, stream>>>(Z, al1, ar1, el, er, nNH);
  k_init<<<(nND + T - 1) / T, T, 0, stream>>>(H32, b1, m, s, nND);
  k_edge_max<<<(nEH + T - 1) / T, T, 0, stream>>>(src, dst, el, er, eb, m, nEH);
  k_edge_expsum<<<(nEH + T - 1) / T, T, 0, stream>>>(dst, eb, m, s, nEH);
  k_edge_agg<<<(unsigned)((nEG + T - 1) / T), T, 0, stream>>>(src, dst, eb, s, Z, H32, nEG);
  k_relu_to_f16<<<(nND + T - 1) / T, T, 0, stream>>>(H32, h16, nND);

  // ----- layer 2 -----
  float* out = (float*)d_out;
  k_gemm_wmma<<<(N + 15) / 16, 256, 0, stream>>>(h16, W2s, Z, N);
  k_eler<<<(nNH + T - 1) / T, T, 0, stream>>>(Z, al2, ar2, el, er, nNH);
  k_init<<<(nND + T - 1) / T, T, 0, stream>>>(out, b2, m, s, nND);
  k_edge_max<<<(nEH + T - 1) / T, T, 0, stream>>>(src, dst, el, er, eb, m, nEH);
  k_edge_expsum<<<(nEH + T - 1) / T, T, 0, stream>>>(dst, eb, m, s, nEH);
  k_edge_agg<<<(unsigned)((nEG + T - 1) / T), T, 0, stream>>>(src, dst, eb, s, Z, out, nEG);
}
